// RecurrentFormulationNet_21784074126007
// MI455X (gfx1250) — compile-verified, matrix operand order checked by reading, not run on hardware
//
#include <hip/hip_runtime.h>

// ---------------- types ----------------
typedef __bf16 bf16_t;
typedef __attribute__((ext_vector_type(16))) __bf16 v16bf;
typedef __attribute__((ext_vector_type(8)))  __bf16 v8bf;
typedef __attribute__((ext_vector_type(4)))  __bf16 v4bf;
typedef __attribute__((ext_vector_type(8)))  float  v8f;
typedef __attribute__((ext_vector_type(4)))  float  v4f;

// ---------------- problem constants (match reference) ----------------
#define NN   30000      // nodes
#define EE   480000     // edges
#define NF   4
#define NM   8
#define HH   128
#define TT   4
#define DMID 8
#define EPSV 1e-5f
#define DTV  0.1f

// ======================================================================
// Setup kernels: degree, scan, CSR build (deterministic), weight convert
// ======================================================================
__global__ void k_zero_int(int* p, int n) {
  int i = blockIdx.x * blockDim.x + threadIdx.x;
  if (i < n) p[i] = 0;
}

__global__ void k_degree(const int* __restrict__ col, int* counts, int n_edges) {
  int e = blockIdx.x * blockDim.x + threadIdx.x;
  if (e < n_edges) atomicAdd(&counts[col[e]], 1);
}

// exclusive scan of (counts[i] + 1) -> rowptr[0..n], single workgroup
__global__ void k_scan(const int* __restrict__ counts, int* rowptr, int n) {
  __shared__ int buf[1024];
  __shared__ int carry;
  int tid = threadIdx.x;
  if (tid == 0) carry = 0;
  __syncthreads();
  for (int base = 0; base < n; base += 1024) {
    int v = (base + tid < n) ? counts[base + tid] + 1 : 0;  // +1 self loop
    buf[tid] = v;
    __syncthreads();
    for (int off = 1; off < 1024; off <<= 1) {
      int t = (tid >= off) ? buf[tid - off] : 0;
      __syncthreads();
      buf[tid] += t;
      __syncthreads();
    }
    if (base + tid < n) rowptr[base + tid] = carry + buf[tid] - v;
    __syncthreads();
    if (tid == 0) carry += buf[1023];
    __syncthreads();
  }
  if (tid == 0) rowptr[n] = carry;
}

__global__ void k_dinv(const int* __restrict__ counts, float* dinv, int n) {
  int i = blockIdx.x * blockDim.x + threadIdx.x;
  if (i < n) dinv[i] = rsqrtf((float)(counts[i] + 1));  // deg >= 1 (self loop)
}

__global__ void k_fill(const int* __restrict__ row, const int* __restrict__ col,
                       const int* __restrict__ rowptr, int* fill, int* srcidx,
                       int n_edges, int n_nodes) {
  int e = blockIdx.x * blockDim.x + threadIdx.x;
  if (e < n_edges) {
    int c = col[e];
    int pos = rowptr[c] + atomicAdd(&fill[c], 1);
    srcidx[pos] = row[e];
  } else if (e < n_edges + n_nodes) {
    int i = e - n_edges;
    int pos = rowptr[i] + atomicAdd(&fill[i], 1);
    srcidx[pos] = i;  // self loop
  }
}

// per-node insertion sort -> CSR order independent of atomic race order
// (makes fp32 edge-sum order deterministic across calls)
__global__ void k_sort(const int* __restrict__ rowptr, int* srcidx, int n) {
  int i = blockIdx.x * blockDim.x + threadIdx.x;
  if (i >= n) return;
  int s = rowptr[i], e = rowptr[i + 1];
  for (int a = s + 1; a < e; a++) {
    int v = srcidx[a];
    int b = a - 1;
    while (b >= s && srcidx[b] > v) { srcidx[b + 1] = srcidx[b]; b--; }
    srcidx[b + 1] = v;
  }
}

// transpose + convert weight: W fp32 [Kin x Hdim] -> Wt bf16 [Hp x Kp] (zero pad)
__global__ void k_wt(const float* __restrict__ W, bf16_t* Wt,
                     int Kin, int Hdim, int Kp, int Hp) {
  int idx = blockIdx.x * blockDim.x + threadIdx.x;
  if (idx >= Hp * Kp) return;
  int h = idx / Kp, k = idx % Kp;
  float v = (k < Kin && h < Hdim) ? W[k * Hdim + h] : 0.0f;
  Wt[idx] = (bf16_t)v;
}

// ======================================================================
// Aggregation: y[i,:] = bf16( dinv[i] * sum_{s in CSR[i]} dinv[s]*act(x[s,:]) )
// one wave per node, lane owns 4 channels (up to 2 passes -> 256 ch max)
// ======================================================================
__global__ void k_agg(const float* __restrict__ x, const int* __restrict__ rowptr,
                      const int* __restrict__ srcidx, const float* __restrict__ dinv,
                      bf16_t* __restrict__ y, int Kin, int Kp, int n, int relu) {
  int node = (blockIdx.x * blockDim.x + threadIdx.x) >> 5;
  int lane = threadIdx.x & 31;
  if (node >= n) return;
  int s0 = rowptr[node], s1 = rowptr[node + 1];
  float acc[2][4] = {};
  for (int e = s0; e < s1; e++) {
    int s = srcidx[e];
    float w = dinv[s];
    const float* xp = x + (long)s * Kin;
#pragma unroll
    for (int p = 0; p < 2; p++) {
      int c = lane * 4 + p * 128;
      if (c < Kin) {
        v4f v = *(const v4f*)(xp + c);
        if (relu) {
          v.x = fmaxf(v.x, 0.f); v.y = fmaxf(v.y, 0.f);
          v.z = fmaxf(v.z, 0.f); v.w = fmaxf(v.w, 0.f);
        }
        acc[p][0] += w * v.x; acc[p][1] += w * v.y;
        acc[p][2] += w * v.z; acc[p][3] += w * v.w;
      }
    }
  }
  float di = dinv[node];
  bf16_t* yp = y + (long)node * Kp;
#pragma unroll
  for (int p = 0; p < 2; p++) {
    int c = lane * 4 + p * 128;
    if (c < Kin) {
      v4bf o;
      o[0] = (bf16_t)(di * acc[p][0]); o[1] = (bf16_t)(di * acc[p][1]);
      o[2] = (bf16_t)(di * acc[p][2]); o[3] = (bf16_t)(di * acc[p][3]);
      *(v4bf*)(yp + c) = o;
    }
  }
  for (int c = Kin + lane; c < Kp; c += 32) yp[c] = (bf16_t)0.0f;  // K padding
}

// ======================================================================
// GEMM epilogue+bias: out[N,Hout] = Y[N,Kp](bf16) x Wt[Hp,Kp]^T(bf16) + b
// wave computes a 16x16 tile via v_wmma_f32_16x16x32_bf16, f32 accumulate.
// A frag (16-bit A layout): lane half selects kb in {0,8}; two 8-elem chunks
// at k0+kb and k0+kb+16. B pre-transposed so same pattern indexed by column.
// C/D: lane -> N, vgpr r -> M (+8 for upper lane half).
// ======================================================================
__global__ void k_gemm(const bf16_t* __restrict__ Y, const bf16_t* __restrict__ Wt,
                       const float* __restrict__ bias, float* __restrict__ out,
                       int Kp, int Hout) {
  int m0 = blockIdx.x * 16;
  int n0 = (threadIdx.x >> 5) * 16;
  int lane = threadIdx.x & 31;
  int lr = lane & 15;
  int kb = (lane >> 4) * 8;
  v8f c = {0.f, 0.f, 0.f, 0.f, 0.f, 0.f, 0.f, 0.f};
  const bf16_t* ap = Y + (long)(m0 + lr) * Kp + kb;
  const bf16_t* bp = Wt + (long)(n0 + lr) * Kp + kb;
  for (int k0 = 0; k0 < Kp; k0 += 32) {
    v8bf a0 = *(const v8bf*)(ap + k0);
    v8bf a1 = *(const v8bf*)(ap + k0 + 16);
    v8bf b0 = *(const v8bf*)(bp + k0);
    v8bf b1 = *(const v8bf*)(bp + k0 + 16);
    v16bf A, B;
#pragma unroll
    for (int i = 0; i < 8; i++) { A[i] = a0[i]; A[i + 8] = a1[i];
                                  B[i] = b0[i]; B[i + 8] = b1[i]; }
    c = __builtin_amdgcn_wmma_f32_16x16x32_bf16(false, A, false, B,
                                                (short)0, c, false, false);
  }
  int ncol = n0 + lr;
  if (ncol < Hout) {
    float bb = bias[ncol];
    int mbase = m0 + ((lane >> 4) << 3);
#pragma unroll
    for (int r = 0; r < 8; r++)
      out[(long)(mbase + r) * Hout + ncol] = c[r] + bb;
  }
}

// ======================================================================
// InstanceNorm (affine=False): one workgroup per channel, fp32, in place.
// Deterministic tree reduction (no atomics).
// ======================================================================
__global__ void k_inorm(float* x, int n, int H) {
  int c = blockIdx.x, tid = threadIdx.x;
  __shared__ float s1[256], s2[256];
  float a = 0.f, b = 0.f;
  for (int i = tid; i < n; i += 256) {
    float v = x[(long)i * H + c];
    a += v; b += v * v;
  }
  s1[tid] = a; s2[tid] = b;
  __syncthreads();
  for (int off = 128; off > 0; off >>= 1) {
    if (tid < off) { s1[tid] += s1[tid + off]; s2[tid] += s2[tid + off]; }
    __syncthreads();
  }
  float m = s1[0] / (float)n;
  float var = s2[0] / (float)n - m * m;
  float rs = rsqrtf(var + EPSV);
  for (int i = tid; i < n; i += 256) {
    long idx = (long)i * H + c;
    x[idx] = (x[idx] - m) * rs;
  }
}

// ---------------- small elementwise kernels ----------------
__global__ void k_tanh(const float* __restrict__ in, float* out, int n) {
  int i = blockIdx.x * blockDim.x + threadIdx.x;
  if (i < n) out[i] = tanhf(in[i]);
}
__global__ void k_copy(const float* __restrict__ in, float* out, int n) {
  int i = blockIdx.x * blockDim.x + threadIdx.x;
  if (i < n) out[i] = in[i];
}
__global__ void k_concat(const float* __restrict__ fcur, const float* __restrict__ mesh,
                         float* xc, int n) {
  int idx = blockIdx.x * blockDim.x + threadIdx.x;
  if (idx >= n * (NF + HH)) return;
  int node = idx / (NF + HH), c = idx % (NF + HH);
  xc[idx] = (c < NF) ? fcur[node * NF + c] : mesh[(long)node * HH + (c - NF)];
}
// F_dot = tanh(g); F_next = tanh(F_cur + F_dot*DT); write [n,T,f] outputs
__global__ void k_finalize(const float* __restrict__ g, float* fcur,
                           float* out, int n, int t) {
  int idx = blockIdx.x * blockDim.x + threadIdx.x;
  if (idx >= n * NF) return;
  int node = idx / NF, f = idx % NF;
  float fdot = tanhf(g[idx]);
  float fnext = tanhf(fcur[idx] + fdot * DTV);
  out[(long)node * (TT * NF) + t * NF + f] = fnext;                      // Fs
  out[(long)n * TT * NF + (long)node * (TT * NF) + t * NF + f] = fdot;   // F_dots
  fcur[idx] = fnext;
}

// ======================================================================
extern "C" void kernel_launch(void* const* d_in, const int* in_sizes, int n_in,
                              void* d_out, int out_size, void* d_ws, size_t ws_size,
                              hipStream_t stream) {
  const float* F_0      = (const float*)d_in[0];
  const int*   eidx     = (const int*)d_in[1];   // [2, E]
  const float* meshf    = (const float*)d_in[2];
  const float* mesh_W0  = (const float*)d_in[4];
  const float* mesh_b0  = (const float*)d_in[5];
  const float* mesh_Wh  = (const float*)d_in[6];
  const float* mesh_bh  = (const float*)d_in[7];
  const float* mesh_W9  = (const float*)d_in[8];
  const float* mesh_b9  = (const float*)d_in[9];
  const float* diff_W0  = (const float*)d_in[10];
  const float* diff_b0  = (const float*)d_in[11];
  const float* diff_Wh  = (const float*)d_in[12];
  const float* diff_bh  = (const float*)d_in[13];
  const float* diff_W9  = (const float*)d_in[14];
  const float* diff_b9  = (const float*)d_in[15];
  float* out = (float*)d_out;

  const int* row = eidx;
  const int* col = eidx + EE;

  // ---- carve workspace ----
  char* base = (char*)d_ws;
  size_t off = 0;
  auto alloc = [&](size_t bytes) -> void* {
    void* r = base + off;
    off = (off + bytes + 255) & ~(size_t)255;
    return r;
  };
  int*    counts  = (int*)alloc(NN * 4);
  int*    rowptr  = (int*)alloc((NN + 1) * 4);
  int*    fill    = (int*)alloc(NN * 4);
  int*    srcidx  = (int*)alloc((size_t)(EE + NN) * 4);
  float*  dinv    = (float*)alloc(NN * 4);
  bf16_t* ybf     = (bf16_t*)alloc((size_t)NN * 160 * 2);
  float*  xbuf    = (float*)alloc((size_t)NN * HH * 4);
  float*  xc      = (float*)alloc((size_t)NN * (NF + HH) * 4);
  float*  meshb   = (float*)alloc((size_t)NN * HH * 4);
  float*  fcur    = (float*)alloc(NN * NF * 4);
  float*  gbuf    = (float*)alloc(NN * NF * 4);
  bf16_t* wt_m0   = (bf16_t*)alloc(128 * 32 * 2);
  bf16_t* wt_mh   = (bf16_t*)alloc(8 * 128 * 128 * 2);
  bf16_t* wt_m9   = (bf16_t*)alloc(128 * 128 * 2);
  bf16_t* wt_d0   = (bf16_t*)alloc(128 * 160 * 2);
  bf16_t* wt_dh   = (bf16_t*)alloc(8 * 128 * 128 * 2);
  bf16_t* wt_d9   = (bf16_t*)alloc(16 * 128 * 2);

  // ---- graph preprocessing (fresh every call; deterministic) ----
  k_zero_int<<<(NN + 255) / 256, 256, 0, stream>>>(counts, NN);
  k_zero_int<<<(NN + 255) / 256, 256, 0, stream>>>(fill, NN);
  k_degree<<<(EE + 255) / 256, 256, 0, stream>>>(col, counts, EE);
  k_scan<<<1, 1024, 0, stream>>>(counts, rowptr, NN);
  k_dinv<<<(NN + 255) / 256, 256, 0, stream>>>(counts, dinv, NN);
  k_fill<<<(EE + NN + 255) / 256, 256, 0, stream>>>(row, col, rowptr, fill, srcidx, EE, NN);
  k_sort<<<(NN + 255) / 256, 256, 0, stream>>>(rowptr, srcidx, NN);

  // ---- weight transpose+convert to bf16 [Hp x Kp] ----
  auto wtc = [&](const float* W, bf16_t* Wt, int Kin, int Hdim, int Kp, int Hp) {
    int tot = Hp * Kp;
    k_wt<<<(tot + 255) / 256, 256, 0, stream>>>(W, Wt, Kin, Hdim, Kp, Hp);
  };
  wtc(mesh_W0, wt_m0, NM, HH, 32, 128);
  for (int l = 0; l < DMID; l++)
    wtc(mesh_Wh + l * HH * HH, wt_mh + l * HH * HH, HH, HH, 128, 128);
  wtc(mesh_W9, wt_m9, HH, HH, 128, 128);
  wtc(diff_W0, wt_d0, NF + HH, HH, 160, 128);
  for (int l = 0; l < DMID; l++)
    wtc(diff_Wh + l * HH * HH, wt_dh + l * HH * HH, HH, HH, 128, 128);
  wtc(diff_W9, wt_d9, HH, NF, 128, 16);

  const int aggBlocks  = (NN + 7) / 8;   // 8 waves (nodes) per 256-thread block
  const int gemmBlocks = NN / 16;        // 30000 % 16 == 0

  auto conv = [&](const float* xin, int Kin, int Kp, int relu,
                  const bf16_t* Wt, const float* b, float* xout, int Hout, int Hp) {
    k_agg<<<aggBlocks, 256, 0, stream>>>(xin, rowptr, srcidx, dinv, ybf, Kin, Kp, NN, relu);
    k_gemm<<<gemmBlocks, 32 * (Hp / 16), 0, stream>>>(ybf, Wt, b, xout, Kp, Hout);
  };

  // ---- mesh descriptor GCN block ----
  conv(meshf, NM, 32, 0, wt_m0, mesh_b0, xbuf, HH, 128);
  k_inorm<<<HH, 256, 0, stream>>>(xbuf, NN, HH);
  for (int l = 0; l < DMID; l++)
    conv(xbuf, HH, 128, 1, wt_mh + l * HH * HH, mesh_bh + l * HH, xbuf, HH, 128);
  conv(xbuf, HH, 128, 0, wt_m9, mesh_b9, xbuf, HH, 128);
  k_tanh<<<((NN * HH) + 255) / 256, 256, 0, stream>>>(xbuf, meshb, NN * HH);

  // ---- time loop: differentiator GCN block ----
  k_copy<<<(NN * NF + 255) / 256, 256, 0, stream>>>(F_0, fcur, NN * NF);
  for (int t = 0; t < TT; t++) {
    k_concat<<<((NN * (NF + HH)) + 255) / 256, 256, 0, stream>>>(fcur, meshb, xc, NN);
    conv(xc, NF + HH, 160, 0, wt_d0, diff_b0, xbuf, HH, 128);
    k_inorm<<<HH, 256, 0, stream>>>(xbuf, NN, HH);
    for (int l = 0; l < DMID; l++)
      conv(xbuf, HH, 128, 1, wt_dh + l * HH * HH, diff_bh + l * HH, xbuf, HH, 128);
    conv(xbuf, HH, 128, 0, wt_d9, diff_b9, gbuf, NF, 16);
    k_finalize<<<(NN * NF + 255) / 256, 256, 0, stream>>>(gbuf, fcur, out, NN, t);
  }
  (void)in_sizes; (void)n_in; (void)out_size; (void)ws_size;
}
